// YOLOv8Loss_86603720556712
// MI455X (gfx1250) — compile-verified, matrix operand order checked by reading, not run on hardware
//
#include <hip/hip_runtime.h>

typedef __attribute__((ext_vector_type(16))) _Float16 v16h;
typedef __attribute__((ext_vector_type(8)))  float    v8f;

#define NCLS   80
#define NBINS  16
#define NANCH  8400
#define MAXGT  32
#define NBATCH 32

__device__ __forceinline__ float wave_reduce_add(float v) {
#pragma unroll
  for (int off = 16; off > 0; off >>= 1)
    v += __shfl_xor(v, off, 32);
  return v;
}

__global__ void yolo_init_kernel(float* __restrict__ acc) {
  if (threadIdx.x < 4) acc[threadIdx.x] = 0.0f;
}

// One wave handles 16 anchors of one scale for one batch image.
// HW/W/STRIDE are compile-time so every channel access is base + 24-bit
// immediate offset (single VGPR address, clauseable loads).
template <int HW, int W, int STRIDE, int AOFF>
__global__ __launch_bounds__(256, 4)
void yolo_scale_kernel(const float* __restrict__ p,
                       const float* __restrict__ gt_bboxes,
                       const int*   __restrict__ gt_labels,
                       const int*   __restrict__ matched,
                       float* __restrict__ acc)
{
  constexpr int TPB = HW / 16;            // tiles per batch image
  __shared__ float s_sum[8][4][16];       // [wave][side][anchor-in-tile]
  __shared__ float s_num[8][4][16];

  const int lane = threadIdx.x & 31;
  const int wv   = threadIdx.x >> 5;
  const int wgid = blockIdx.x * 8 + wv;   // 0 .. NBATCH*TPB-1
  const int b    = wgid / TPB;            // constant divide -> mulhi
  const int tile = wgid - b * TPB;
  const int al0  = tile * 16;             // local anchor base within scale

  const int row = lane & 15;              // anchor row within tile (A/D matrix row)
  const int hi  = lane >> 4;              // which half of the 16 bins this lane carries
  const int al  = al0 + row;              // local anchor index within scale
  const float* pb = p + (size_t)b * 144 * HW + al;   // channel c at pb[c*HW] (imm offset)

  // Prefetch the class-logit region for this lane (global_prefetch_b8).
  __builtin_prefetch(pb + (size_t)64 * HW, 0, 0);

  // ---------------- Stage 1: DFL softmax sums via WMMA ----------------
  // A (16x32 f16): row = anchor, K = bin (K=16..31 zero pad).
  //   lane hi=0 carries K=0..7 in elements 0..7; lane hi=1 carries K=8..15.
  // B = all-ones f16 (layout-invariant), so D[m][*] = sum_k A[m][k].
  v16h bOnes;
#pragma unroll
  for (int i = 0; i < 16; ++i) bOnes[i] = (_Float16)1.0f;
  v8f czero = {};

#pragma unroll
  for (int s = 0; s < 4; ++s) {
    float ex[8];
#pragma unroll
    for (int j = 0; j < 8; ++j)
      ex[j] = __expf(pb[(size_t)(s * NBINS + hi * 8 + j) * HW]);

    v16h aExp, aWgt;
#pragma unroll
    for (int j = 0; j < 8; ++j) {
      aExp[j]     = (_Float16)ex[j];
      aWgt[j]     = (_Float16)(ex[j] * (float)(hi * 8 + j));   // exp * bin-index
      aExp[j + 8] = (_Float16)0.0f;
      aWgt[j + 8] = (_Float16)0.0f;
    }

    v8f dsum = __builtin_amdgcn_wmma_f32_16x16x32_f16(false, aExp, false, bOnes,
                                                      (short)0, czero, false, false);
    v8f dnum = __builtin_amdgcn_wmma_f32_16x16x32_f16(false, aWgt, false, bOnes,
                                                      (short)0, czero, false, false);
    // D layout: VGPR r holds M=r (lanes 0-15) / M=8+r (lanes 16-31); all cols equal.
    if (row == 0) {
#pragma unroll
      for (int r = 0; r < 8; ++r) {
        s_sum[wv][s][hi * 8 + r] = dsum[r];
        s_num[wv][s][hi * 8 + r] = dnum[r];
      }
    }
  }
  __syncthreads();

  // ---------------- Stage 2: per-anchor epilogue ----------------
  const int  ga  = AOFF + al;                      // global anchor id
  const int  mi  = matched[b * NANCH + ga];
  const bool pos = (mi >= 0);
  const int  idx = pos ? mi : 0;
  const int  gtb = b * MAXGT + idx;
  const float tx1 = gt_bboxes[gtb * 4 + 0];
  const float ty1 = gt_bboxes[gtb * 4 + 1];
  const float tx2 = gt_bboxes[gtb * 4 + 2];
  const float ty2 = gt_bboxes[gtb * 4 + 3];
  const int   lbl = gt_labels[gtb];

  const int   h  = al / W, w = al - h * W;         // constant W -> magic multiply
  const float cx = ((float)w + 0.5f) * (float)STRIDE;
  const float cy = ((float)h + 0.5f) * (float)STRIDE;

  float td[4];
  td[0] = fmaxf(cx - tx1, 0.0f);
  td[1] = fmaxf(cy - ty1, 0.0f);
  td[2] = fmaxf(tx2 - cx, 0.0f);
  td[3] = fmaxf(ty2 - cy, 0.0f);
  constexpr float inv_stride = 1.0f / (float)STRIDE;

  // DFL loss: lane-half handles sides {2*hi, 2*hi+1} of its anchor.
  float dfl_part = 0.0f;
#pragma unroll
  for (int k = 0; k < 2; ++k) {
    const int s = hi * 2 + k;
    float tb = fminf(fmaxf(td[s] * inv_stride, 0.0f), (float)(NBINS - 1) - 1e-6f);
    const int lo  = (int)tb;
    const int hb  = min(lo + 1, NBINS - 1);
    const float a = tb - (float)lo;
    const float L = __logf(s_sum[wv][s][row]);                 // log(sum exp)
    const float xlo = pb[(size_t)(s * NBINS + lo) * HW];       // L2-hot re-reads
    const float xhi = pb[(size_t)(s * NBINS + hb) * HW];
    dfl_part += L - (1.0f - a) * xlo - a * xhi;
  }
  dfl_part = pos ? dfl_part : 0.0f;

  // IoU loss + positive count: hi==0 half only (once per anchor).
  float iou_part = 0.0f, cnt_part = 0.0f;
  if (hi == 0) {
    const float d0 = s_num[wv][0][row] / s_sum[wv][0][row] * (float)STRIDE;
    const float d1 = s_num[wv][1][row] / s_sum[wv][1][row] * (float)STRIDE;
    const float d2 = s_num[wv][2][row] / s_sum[wv][2][row] * (float)STRIDE;
    const float d3 = s_num[wv][3][row] / s_sum[wv][3][row] * (float)STRIDE;
    const float px1 = cx - d0, py1 = cy - d1, px2 = cx + d2, py2 = cy + d3;
    const float ix1 = fmaxf(px1, tx1), iy1 = fmaxf(py1, ty1);
    const float ix2 = fminf(px2, tx2), iy2 = fminf(py2, ty2);
    const float inter = fmaxf(ix2 - ix1, 0.0f) * fmaxf(iy2 - iy1, 0.0f);
    const float ap = fmaxf(px2 - px1, 0.0f) * fmaxf(py2 - py1, 0.0f);
    const float at = fmaxf(tx2 - tx1, 0.0f) * fmaxf(ty2 - ty1, 0.0f);
    const float iou = inter / (ap + at - inter + 1e-7f);
    iou_part = pos ? (1.0f - iou) : 0.0f;
    cnt_part = pos ? 1.0f : 0.0f;
  }

  // Focal classification loss: lane-half handles classes [hi*40, hi*40+40).
  // One exp per logit: e = exp(-|s|) drives both sigmoid and log1p.
  float cls_part = 0.0f;
#pragma unroll 8
  for (int c = 0; c < 40; ++c) {
    const int   cg = hi * 40 + c;
    const float sv = pb[(size_t)(64 + cg) * HW];
    const float t  = (pos && (lbl == cg)) ? 1.0f : 0.0f;
    const float e  = __expf(-fabsf(sv));
    const float q  = e / (1.0f + e);                 // sigmoid(-|sv|)
    const float pr = (sv >= 0.0f) ? (1.0f - q) : q;  // sigmoid(sv)
    const float ce = fmaxf(sv, 0.0f) - sv * t + __logf(1.0f + e);
    const float om = t * (1.0f - pr) + (1.0f - t) * pr;   // 1 - p_t
    const float af = t * 0.25f + (1.0f - t) * 0.75f;
    cls_part += af * om * om * ce;
  }

  // Wave reduction + one atomic per wave per accumulator.
  const float r0 = wave_reduce_add(cls_part);
  const float r1 = wave_reduce_add(iou_part);
  const float r2 = wave_reduce_add(dfl_part);
  const float r3 = wave_reduce_add(cnt_part);
  if (lane == 0) {
    atomicAdd(&acc[0], r0);
    atomicAdd(&acc[1], r1);
    atomicAdd(&acc[2], r2);
    atomicAdd(&acc[3], r3);
  }
}

__global__ void yolo_fin_kernel(const float* __restrict__ acc, float* __restrict__ out) {
  const float np = fmaxf(acc[3], 1.0f);
  out[0] = 1.0f * acc[0] / np + 7.5f * acc[1] / np + 1.5f * acc[2] / (np * 4.0f);
}

extern "C" void kernel_launch(void* const* d_in, const int* in_sizes, int n_in,
                              void* d_out, int out_size, void* d_ws, size_t ws_size,
                              hipStream_t stream) {
  const float* p0  = (const float*)d_in[0];
  const float* p1  = (const float*)d_in[1];
  const float* p2  = (const float*)d_in[2];
  const float* gtb = (const float*)d_in[3];
  const int*   gtl = (const int*)d_in[4];
  const int*   mi  = (const int*)d_in[5];
  float* acc = (float*)d_ws;            // 4 floats of scratch
  float* out = (float*)d_out;

  yolo_init_kernel<<<1, 32, 0, stream>>>(acc);
  // scale 0: 80x80, 32*400 tiles -> 12800 waves -> 1600 blocks
  yolo_scale_kernel<6400, 80,  8,    0><<<1600, 256, 0, stream>>>(p0, gtb, gtl, mi, acc);
  // scale 1: 40x40, 32*100 tiles -> 3200 waves -> 400 blocks
  yolo_scale_kernel<1600, 40, 16, 6400><<< 400, 256, 0, stream>>>(p1, gtb, gtl, mi, acc);
  // scale 2: 20x20, 32*25 tiles -> 800 waves -> 100 blocks
  yolo_scale_kernel< 400, 20, 32, 8000><<< 100, 256, 0, stream>>>(p2, gtb, gtl, mi, acc);
  yolo_fin_kernel<<<1, 1, 0, stream>>>(acc, out);
}
